// GPT_49795850830101
// MI455X (gfx1250) — matrix operand order repaired and checked by reading, not compiled, not run on hardware
//
#include <hip/hip_runtime.h>
#include <math.h>

// ---------------------------------------------------------------------------
// Types for CDNA5 WMMA (gfx1250, wave32)
// ---------------------------------------------------------------------------
typedef __attribute__((ext_vector_type(16))) _Float16 v16h;
typedef __attribute__((ext_vector_type(8)))  _Float16 v8h;
typedef __attribute__((ext_vector_type(4)))  _Float16 v4h;
typedef __attribute__((ext_vector_type(8)))  float    v8f;

union AFrag {
  v16h v;
  v8h  h[2];
};

__device__ __forceinline__ v8f wmma_f16(v16h a, v16h b, v8f c) {
  // (neg_a, A, neg_b, B, c_mod, C, reuse_a, reuse_b)
  return __builtin_amdgcn_wmma_f32_16x16x32_f16(false, a, false, b, (short)0, c,
                                                false, false);
}

// ---------------------------------------------------------------------------
// Embedding gather: x[t,:] = emb[tokens[t],:]
// ---------------------------------------------------------------------------
__global__ void embed_kernel(const int* __restrict__ tokens,
                             const float* __restrict__ emb,
                             float* __restrict__ x, int d) {
  int t = blockIdx.x;
  int tok = tokens[t];
  const float* src = emb + (size_t)tok * d;
  float* dst = x + (size_t)t * d;
  for (int i = threadIdx.x; i < d; i += blockDim.x) dst[i] = src[i];
}

// ---------------------------------------------------------------------------
// RMSNorm: y[row,:] = w * x[row,:] * rsqrt(mean(x^2)+eps)
// ---------------------------------------------------------------------------
__global__ __launch_bounds__(256) void rmsnorm_kernel(
    const float* __restrict__ x, const float* __restrict__ w,
    float* __restrict__ y, int d) {
  int row = blockIdx.x;
  const float* xr = x + (size_t)row * d;
  float s = 0.f;
  for (int i = threadIdx.x; i < d; i += blockDim.x) {
    float v = xr[i];
    s += v * v;
  }
  __shared__ float red[8];
  #pragma unroll
  for (int o = 16; o > 0; o >>= 1) s += __shfl_down(s, o, 32);
  if ((threadIdx.x & 31) == 0) red[threadIdx.x >> 5] = s;
  __syncthreads();
  __shared__ float rstd;
  if (threadIdx.x == 0) {
    float t = 0.f;
    #pragma unroll
    for (int i = 0; i < 8; ++i) t += red[i];
    rstd = rsqrtf(t / (float)d + 1e-8f);
  }
  __syncthreads();
  float r = rstd;
  float* yr = y + (size_t)row * d;
  for (int i = threadIdx.x; i < d; i += blockDim.x) yr[i] = w[i] * xr[i] * r;
}

// ---------------------------------------------------------------------------
// RoPE in-place on q and k slices of qkv buffer (row = token, width 3d)
// ---------------------------------------------------------------------------
__global__ void rope_kernel(float* __restrict__ qkv, int L, int d) {
  int t = blockIdx.x;          // global token (B*L)
  int pos = t % L;
  float* row = qkv + (size_t)t * 3 * d;
  int halfd = d >> 1;
  for (int i = threadIdx.x; i < d; i += blockDim.x) {  // d pairs: q then k
    int which = i / halfd;     // 0 = q, 1 = k
    int p = i % halfd;         // pair index
    float* base = row + which * d;
    float th = __powf(10000.0f, (-2.0f * (float)p) / (float)d);
    float ang = (float)pos * th;
    float sn = __sinf(ang), cs = __cosf(ang);
    float x0 = base[2 * p], x1 = base[2 * p + 1];
    base[2 * p]     = x0 * cs - x1 * sn;
    base[2 * p + 1] = x1 * cs + x0 * sn;
  }
}

// ---------------------------------------------------------------------------
// SiLU(u) * v elementwise
// ---------------------------------------------------------------------------
__global__ void silumul_kernel(const float* __restrict__ u,
                               const float* __restrict__ v,
                               float* __restrict__ g, long long n) {
  long long i = (long long)blockIdx.x * blockDim.x + threadIdx.x;
  if (i < n) {
    float a = u[i];
    float s = a / (1.f + __expf(-a));
    g[i] = s * v[i];
  }
}

// ---------------------------------------------------------------------------
// WMMA GEMM: C[M,N] = A[M,K] @ W[N,K]^T (+bias[N]) (+resid[M,N])
// fp32 in memory, f16 operands in LDS, f32 WMMA accumulate.
// Block: 256 threads = 8 waves (4 along M x 2 along N), BM=128 BN=64 BK=32.
// Double-buffered LDS: global loads for chunk i+1 issue before the WMMAs of
// chunk i; all guards are branch-free (clamped address + cndmask-zero) so the
// weight stream runs as batched global_load_b128 with full MLP.
// Grid: (M/128, ceil(N/64)) -- M fastest so resident blocks share W in L2.
// ---------------------------------------------------------------------------
#define GBM 128
#define GBN 64
#define GBK 32
#define GLDS 40  // halfs per LDS row (32 data + 8 pad, keeps rows 16B aligned)

__global__ __launch_bounds__(256) void gemm_wmma_kernel(
    const float* __restrict__ A, const float* __restrict__ W,
    const float* __restrict__ bias, const float* __restrict__ resid,
    float* __restrict__ C, int M, int N, int K) {
  __shared__ _Float16 As[2][GBM * GLDS];
  __shared__ _Float16 Bs[2][GBN * GLDS];

  const int tid = threadIdx.x;
  const int wave = tid >> 5, lane = tid & 31;
  const int half = lane >> 4, l16 = lane & 15;
  const int wm = wave & 3, wn = wave >> 2;
  const int m0 = blockIdx.x * GBM, n0 = blockIdx.y * GBN;
  const int lrid = tid >> 3;  // 0..31 (row within pass)
  const int lcid = tid & 7;   // 0..7  (4-col group)

  const bool vec4 = ((K & 3) == 0);  // uniform: float4 path legal
  const int K32 = K & ~(GBK - 1);

  v8f acc[2][2] = {};

  // ---- chunk loaders: unconditional loads, branch-free guards --------------
  auto loadA = [&](int k0, float4* fa) {
    #pragma unroll
    for (int p = 0; p < 4; ++p) {
      const float* src = A + (size_t)(m0 + lrid + p * 32) * K + k0 + lcid * 4;
      if (vec4) {
        fa[p] = *(const float4*)src;
      } else {
        fa[p].x = src[0]; fa[p].y = src[1]; fa[p].z = src[2]; fa[p].w = src[3];
      }
    }
  };
  auto loadB = [&](int k0, float4* fb) {
    #pragma unroll
    for (int p = 0; p < 2; ++p) {
      int gn = n0 + lrid + p * 32;
      int gncl = gn < N ? gn : N - 1;  // clamp address, zero value below
      const float* src = W + (size_t)gncl * K + k0 + lcid * 4;
      float4 f;
      if (vec4) {
        f = *(const float4*)src;
      } else {
        f.x = src[0]; f.y = src[1]; f.z = src[2]; f.w = src[3];
      }
      if (gn >= N) { f.x = 0.f; f.y = 0.f; f.z = 0.f; f.w = 0.f; }
      fb[p] = f;
    }
  };
  auto storeLDS = [&](int buf, const float4* fa, const float4* fb) {
    #pragma unroll
    for (int p = 0; p < 4; ++p) {
      v4h hv = {(_Float16)fa[p].x, (_Float16)fa[p].y, (_Float16)fa[p].z,
                (_Float16)fa[p].w};
      *(v4h*)&As[buf][(lrid + p * 32) * GLDS + lcid * 4] = hv;
    }
    #pragma unroll
    for (int p = 0; p < 2; ++p) {
      v4h hv = {(_Float16)fb[p].x, (_Float16)fb[p].y, (_Float16)fb[p].z,
                (_Float16)fb[p].w};
      *(v4h*)&Bs[buf][(lrid + p * 32) * GLDS + lcid * 4] = hv;
    }
  };
  auto compute = [&](int buf) {
    AFrag a[2], b[2];
    #pragma unroll
    for (int mi = 0; mi < 2; ++mi) {
      const _Float16* base = &As[buf][(wm * 32 + mi * 16 + l16) * GLDS];
      a[mi].h[0] = *(const v8h*)(base + half * 8);       // K {0..7}/{8..15}
      a[mi].h[1] = *(const v8h*)(base + 16 + half * 8);  // K {16..23}/{24..31}
    }
    #pragma unroll
    for (int ni = 0; ni < 2; ++ni) {
      const _Float16* base = &Bs[buf][(wn * 32 + ni * 16 + l16) * GLDS];
      b[ni].h[0] = *(const v8h*)(base + half * 8);       // K {0..7}/{8..15}
      b[ni].h[1] = *(const v8h*)(base + 16 + half * 8);  // K {16..23}/{24..31}
    }
    #pragma unroll
    for (int mi = 0; mi < 2; ++mi)
      #pragma unroll
      for (int ni = 0; ni < 2; ++ni)
        acc[mi][ni] = wmma_f16(a[mi].v, b[ni].v, acc[mi][ni]);
  };

  // ---- software-pipelined main loop over full K-chunks ---------------------
  float4 fa[4], fb[2];
  const int nchunk = K32 / GBK;
  if (nchunk > 0) {
    loadA(0, fa);
    loadB(0, fb);
    storeLDS(0, fa, fb);
    __syncthreads();
    int cur = 0;
    for (int i = 0; i < nchunk; ++i) {
      bool hasNext = (i + 1) < nchunk;
      if (hasNext) {
        loadA((i + 1) * GBK, fa);  // issue next chunk's loads first
        loadB((i + 1) * GBK, fb);
      }
      compute(cur);                // WMMAs overlap in-flight loads
      if (hasNext) {
        storeLDS(1 - cur, fa, fb);
        __syncthreads();
        cur ^= 1;
      }
    }
  }

  // ---- guarded tail chunk (K % 32 != 0, e.g. K = 5461) ---------------------
  if (K32 < K) {
    __syncthreads();
    #pragma unroll
    for (int p = 0; p < 4; ++p) {
      const float* src = A + (size_t)(m0 + lrid + p * 32) * K;
      #pragma unroll
      for (int j = 0; j < 4; ++j) {
        int gc = K32 + lcid * 4 + j;
        int gccl = gc < K ? gc : K - 1;
        float v = src[gccl];
        As[0][(lrid + p * 32) * GLDS + lcid * 4 + j] =
            (_Float16)(gc < K ? v : 0.f);
      }
    }
    #pragma unroll
    for (int p = 0; p < 2; ++p) {
      int gn = n0 + lrid + p * 32;
      int gncl = gn < N ? gn : N - 1;
      const float* src = W + (size_t)gncl * K;
      #pragma unroll
      for (int j = 0; j < 4; ++j) {
        int gc = K32 + lcid * 4 + j;
        int gccl = gc < K ? gc : K - 1;
        float v = src[gccl];
        Bs[0][(lrid + p * 32) * GLDS + lcid * 4 + j] =
            (_Float16)((gc < K && gn < N) ? v : 0.f);
      }
    }
    __syncthreads();
    compute(0);
  }

  // ---- epilogue: C layout VGPR r -> row r (+8 for hi half), col = lane&15
  #pragma unroll
  for (int mi = 0; mi < 2; ++mi) {
    #pragma unroll
    for (int ni = 0; ni < 2; ++ni) {
      #pragma unroll
      for (int r = 0; r < 8; ++r) {
        int row = m0 + wm * 32 + mi * 16 + r + half * 8;
        int col = n0 + wn * 32 + ni * 16 + l16;
        if (row < M && col < N) {
          float v = acc[mi][ni][r];
          if (bias) v += bias[col];
          if (resid) v += resid[(size_t)row * N + col];
          C[(size_t)row * N + col] = v;
        }
      }
    }
  }
}

// ---------------------------------------------------------------------------
// Flash attention, causal.  Block = 256 threads (8 waves), each wave owns a
// 16-row query tile; block covers 128 rows of one (b, h).  dh = 128.
// qkv layout: [B*L, 3d] with q at 0, k at d, v at 2d; per-head slice h*dh.
// ---------------------------------------------------------------------------
#define ADH 128   // head dim
#define AQS 136   // Qs/Ks LDS stride (halfs)
#define AVS 40    // Vt/Ps LDS stride (halfs)

__global__ __launch_bounds__(256) void attn_kernel(
    const float* __restrict__ qkv, float* __restrict__ headout,
    int L, int d, int nh) {
  const int dh = ADH;
  const int b = blockIdx.z, h = blockIdx.y;
  const int m0 = blockIdx.x * 128;
  const int tid = threadIdx.x, wave = tid >> 5, lane = tid & 31;
  const int half = lane >> 4, l16 = lane & 15;
  const float scale = rsqrtf((float)d);  // reference scales by full d

  __shared__ _Float16 Qs[128 * AQS];
  __shared__ _Float16 Ks[32 * AQS];
  __shared__ _Float16 Vt[ADH * AVS];          // transposed: [feat][key]
  __shared__ _Float16 Ps[8 * 16 * AVS];       // per-wave P staging

  const size_t rowstride = (size_t)3 * d;

  // ---- stage Q tile (128 x 128), pre-scaled
  {
    int cid = tid & 31, rid = tid >> 5;
    #pragma unroll
    for (int p = 0; p < 16; ++p) {
      int r = rid + p * 8;
      const float* src =
          qkv + (size_t)(b * L + m0 + r) * rowstride + h * dh + cid * 4;
      float4 f = *(const float4*)src;
      v4h hv = {(_Float16)(f.x * scale), (_Float16)(f.y * scale),
                (_Float16)(f.z * scale), (_Float16)(f.w * scale)};
      *(v4h*)&Qs[r * AQS + cid * 4] = hv;
    }
  }
  __syncthreads();

  // ---- per-wave Q fragments (4 K-chunks of 32 over dh=128)
  AFrag qa[4];
  {
    const _Float16* base = &Qs[(wave * 16 + l16) * AQS];
    #pragma unroll
    for (int kc = 0; kc < 4; ++kc) {
      qa[kc].h[0] = *(const v8h*)(base + kc * 32 + half * 8);
      qa[kc].h[1] = *(const v8h*)(base + kc * 32 + 16 + half * 8);
    }
  }

  v8f o[8] = {};
  float mi[8], li[8];
  #pragma unroll
  for (int r = 0; r < 8; ++r) { mi[r] = -INFINITY; li[r] = 0.f; }

  const int jend = m0 + 128;
  for (int j0 = 0; j0 < jend; j0 += 32) {
    // ---- stage K chunk (32 x 128) and transposed V chunk
    {
      int cid = tid & 31, rid = tid >> 5;
      #pragma unroll
      for (int p = 0; p < 4; ++p) {
        int kr = rid + p * 8;  // key 0..31
        const float* krow =
            qkv + (size_t)(b * L + j0 + kr) * rowstride + d + h * dh;
        float4 f = *(const float4*)(krow + cid * 4);
        v4h hv = {(_Float16)f.x, (_Float16)f.y, (_Float16)f.z, (_Float16)f.w};
        *(v4h*)&Ks[kr * AQS + cid * 4] = hv;
        const float* vrow =
            qkv + (size_t)(b * L + j0 + kr) * rowstride + 2 * d + h * dh;
        float4 g = *(const float4*)(vrow + cid * 4);
        Vt[(cid * 4 + 0) * AVS + kr] = (_Float16)g.x;
        Vt[(cid * 4 + 1) * AVS + kr] = (_Float16)g.y;
        Vt[(cid * 4 + 2) * AVS + kr] = (_Float16)g.z;
        Vt[(cid * 4 + 3) * AVS + kr] = (_Float16)g.w;
      }
    }
    __syncthreads();

    bool active = (m0 + wave * 16 + 15) >= j0;  // wave-uniform
    if (active) {
      // ---- S = (Q*scale) @ K^T  (two 16x16 tiles for 32 keys)
      v8f s0 = {}, s1 = {};
      #pragma unroll
      for (int kc = 0; kc < 4; ++kc) {
        AFrag b0, b1;
        const _Float16* kb0 = &Ks[l16 * AQS + kc * 32];
        b0.h[0] = *(const v8h*)(kb0 + half * 8);
        b0.h[1] = *(const v8h*)(kb0 + 16 + half * 8);
        const _Float16* kb1 = &Ks[(16 + l16) * AQS + kc * 32];
        b1.h[0] = *(const v8h*)(kb1 + half * 8);
        b1.h[1] = *(const v8h*)(kb1 + 16 + half * 8);
        s0 = wmma_f16(qa[kc].v, b0.v, s0);
        s1 = wmma_f16(qa[kc].v, b1.v, s1);
      }

      // ---- causal mask + online softmax (row stats per VGPR index r)
      float al[8];
      #pragma unroll
      for (int r = 0; r < 8; ++r) {
        int grow = m0 + wave * 16 + r + half * 8;
        float v0 = (j0 + l16 <= grow) ? s0[r] : -INFINITY;
        float v1 = (j0 + 16 + l16 <= grow) ? s1[r] : -INFINITY;
        float mx = fmaxf(v0, v1);
        #pragma unroll
        for (int off = 8; off > 0; off >>= 1)
          mx = fmaxf(mx, __shfl_xor(mx, off, 32));
        float mn = fmaxf(mi[r], mx);
        float a = __expf(mi[r] - mn);
        float p0 = __expf(v0 - mn);
        float p1 = __expf(v1 - mn);
        s0[r] = p0;
        s1[r] = p1;
        float rs = p0 + p1;
        #pragma unroll
        for (int off = 8; off > 0; off >>= 1) rs += __shfl_xor(rs, off, 32);
        li[r] = li[r] * a + rs;
        mi[r] = mn;
        al[r] = a;
      }
      #pragma unroll
      for (int nt = 0; nt < 8; ++nt)
        #pragma unroll
        for (int r = 0; r < 8; ++r) o[nt][r] *= al[r];

      // ---- P: C-layout -> A-fragment layout via per-wave LDS strip
      _Float16* pw = &Ps[wave * 16 * AVS];
      #pragma unroll
      for (int r = 0; r < 8; ++r) {
        pw[(r + half * 8) * AVS + l16] = (_Float16)s0[r];
        pw[(r + half * 8) * AVS + 16 + l16] = (_Float16)s1[r];
      }
      AFrag pa;
      {
        const _Float16* base = &Ps[wave * 16 * AVS + l16 * AVS];
        pa.h[0] = *(const v8h*)(base + half * 8);
        pa.h[1] = *(const v8h*)(base + 16 + half * 8);
      }

      // ---- O += P @ V  (8 feature tiles of 16)
      #pragma unroll
      for (int nt = 0; nt < 8; ++nt) {
        AFrag vb;
        const _Float16* vbase = &Vt[(nt * 16 + l16) * AVS + half * 8];
        vb.h[0] = *(const v8h*)(vbase);
        vb.h[1] = *(const v8h*)(vbase + 16);
        o[nt] = wmma_f16(pa.v, vb.v, o[nt]);
      }
    }
    __syncthreads();
  }

  // ---- epilogue: head[b,l, h*dh + feat] = O / l
  float inv[8];
  #pragma unroll
  for (int r = 0; r < 8; ++r) inv[r] = 1.f / li[r];
  #pragma unroll
  for (int nt = 0; nt < 8; ++nt) {
    #pragma unroll
    for (int r = 0; r < 8; ++r) {
      int grow = m0 + wave * 16 + r + half * 8;
      int gcol = h * dh + nt * 16 + l16;
      headout[(size_t)(b * L + grow) * d + gcol] = o[nt][r] * inv[r];
    }
  }
}

// ---------------------------------------------------------------------------
// Host driver
// ---------------------------------------------------------------------------
extern "C" void kernel_launch(void* const* d_in, const int* in_sizes, int n_in,
                              void* d_out, int out_size, void* d_ws,
                              size_t ws_size, hipStream_t stream) {
  const int B = 2, L = 1024, d = 2048, NL = 4, NH = 16, V = 32000;
  const int M = B * L;                         // 2048 tokens
  const int H = in_sizes[9] / (NL * d);        // 5461

  const int* tokens = (const int*)d_in[0];
  const float* emb = (const float*)d_in[1];
  const float* out_w = (const float*)d_in[2];
  const float* wx_w = (const float*)d_in[3];
  const float* wx_b = (const float*)d_in[4];
  const float* wo_w = (const float*)d_in[5];
  const float* wo_b = (const float*)d_in[6];
  const float* mnw = (const float*)d_in[7];
  const float* fnw = (const float*)d_in[8];
  const float* u_w = (const float*)d_in[9];
  const float* v_w = (const float*)d_in[10];
  const float* w_w = (const float*)d_in[11];

  float* ws = (float*)d_ws;
  size_t off = 0;
  float* x = ws + off;    off += (size_t)M * d;       // residual stream
  float* xn = ws + off;   off += (size_t)M * d;       // normed activations
  float* qkv = ws + off;  off += (size_t)M * 3 * d;   // fused qkv
  float* head = ws + off; off += (size_t)M * d;       // attention output
  float* ub = ws + off;   off += (size_t)M * H;       // up-proj u (and g)
  float* vb = ws + off;   off += (size_t)M * H;       // up-proj v
  float* logits = (float*)d_out;

  auto gemm = [&](const float* A, const float* W, const float* bias,
                  const float* resid, float* C, int Mm, int Nn, int Kk) {
    dim3 grid(Mm / GBM, (Nn + GBN - 1) / GBN);
    gemm_wmma_kernel<<<grid, 256, 0, stream>>>(A, W, bias, resid, C, Mm, Nn,
                                               Kk);
  };

  embed_kernel<<<M, 256, 0, stream>>>(tokens, emb, x, d);

  for (int l = 0; l < NL; ++l) {
    const float* lwx = wx_w + (size_t)l * 3 * d * d;
    const float* lwxb = wx_b + (size_t)l * 3 * d;
    const float* lwo = wo_w + (size_t)l * d * d;
    const float* lwob = wo_b + (size_t)l * d;
    const float* lmn = mnw + (size_t)l * d;
    const float* lfn = fnw + (size_t)l * d;
    const float* lu = u_w + (size_t)l * H * d;
    const float* lv = v_w + (size_t)l * H * d;
    const float* lw = w_w + (size_t)l * d * H;

    // attention block
    rmsnorm_kernel<<<M, 256, 0, stream>>>(x, lmn, xn, d);
    gemm(xn, lwx, lwxb, nullptr, qkv, M, 3 * d, d);
    rope_kernel<<<M, 256, 0, stream>>>(qkv, L, d);
    {
      dim3 grid(L / 128, NH, B);
      attn_kernel<<<grid, 256, 0, stream>>>(qkv, head, L, d, NH);
    }
    gemm(head, lwo, lwob, x, x, M, d, d);  // residual add, in-place

    // FFN block
    rmsnorm_kernel<<<M, 256, 0, stream>>>(x, lfn, xn, d);
    gemm(xn, lu, nullptr, nullptr, ub, M, H, d);
    gemm(xn, lv, nullptr, nullptr, vb, M, H, d);
    {
      long long n = (long long)M * H;
      int blocks = (int)((n + 255) / 256);
      silumul_kernel<<<blocks, 256, 0, stream>>>(ub, vb, ub, n);
    }
    gemm(ub, lw, nullptr, x, x, M, d, H);  // residual add, in-place
  }

  // logits
  gemm(x, out_w, nullptr, nullptr, logits, M, V, d);
}
